// MultiHeadAttention_893353198057
// MI455X (gfx1250) — matrix operand order repaired and checked by reading, not compiled, not run on hardware
//
#include <hip/hip_runtime.h>

// ---------------------------------------------------------------------------
// MI455X (gfx1250) multi-head attention, bf16 WMMA pipeline.
//   D_MODEL=1024, H=16, Dh=64, B=4, S=2048, causal.
// Compute-bound (~120 GFLOP vs ~10us of HBM traffic at 23.3 TB/s):
//   pass 0: fp32 -> bf16 conversion of activations + weights (bandwidth noise)
//   pass 1: all-bf16 GEMMs with double-buffered GLOBAL_LOAD_ASYNC_TO_LDS tiles
//   pass 2: flash attention, v_wmma_f32_16x16x32_bf16 for QK^T and PV
//   pass 3: output projection (+bias) in the same async GEMM
// ---------------------------------------------------------------------------

typedef __bf16 bf16;
typedef __attribute__((ext_vector_type(16))) __bf16 v16bf;
typedef __attribute__((ext_vector_type(8)))  __bf16 v8bf;
typedef __attribute__((ext_vector_type(8)))  float  v8f;
typedef __attribute__((ext_vector_type(4)))  float  v4f;

#define D_MODEL 1024
#define N_HEADS 16
#define D_HEAD  64
#define BATCH   4
#define SEQ     2048
#define MROWS   (BATCH * SEQ)          // 8192

// --- CDNA5 async global->LDS path (guarded; falls back to plain copies) ----
#if defined(__has_builtin)
#if __has_builtin(__builtin_amdgcn_global_load_async_to_lds_b128) && \
    __has_builtin(__builtin_amdgcn_s_wait_asynccnt)
#define HAS_ASYNC_LDS 1
#endif
#endif
#ifndef HAS_ASYNC_LDS
#define HAS_ASYNC_LDS 0
#endif

// Builtin signature (from hipcc diagnostic): first param is a pointer to a
// 16-byte int vector in the global (AS1 / __device__) address space; second
// is the LDS (AS3 / __shared__) counterpart; then imm offset and imm cpol.
typedef int v4i __attribute__((vector_size(16)));
typedef __attribute__((address_space(1))) v4i* gas_v4i;
typedef __attribute__((address_space(3))) v4i* las_v4i;

__device__ __forceinline__ void lds_copy16(const bf16* g, bf16* l) {
#if HAS_ASYNC_LDS
  // Generic->AS(3) via integer: ISA 10.2 maps LDS aperture addr via addr[31:0].
  __builtin_amdgcn_global_load_async_to_lds_b128(
      (gas_v4i)(unsigned long long)(uintptr_t)g,
      (las_v4i)(unsigned int)(uintptr_t)l, 0, 0);
#else
  *(v8bf*)l = *(const v8bf*)g;
#endif
}

template <int N>
__device__ __forceinline__ void wait_async_lds() {
#if HAS_ASYNC_LDS
  __builtin_amdgcn_s_wait_asynccnt(N);
#endif
}

__device__ __forceinline__ v8f wmma_bf16(v16bf a, v16bf b, v8f c) {
  // 8 args: (neg_a, A, neg_b, B, c_mod, C, reuse_a, reuse_b)
  return __builtin_amdgcn_wmma_f32_16x16x32_bf16(false, a, false, b,
                                                 (short)0, c, false, false);
}

__device__ __forceinline__ v16bf cat8(v8bf a, v8bf b) {
  return __builtin_shufflevector(a, b, 0,1,2,3,4,5,6,7,8,9,10,11,12,13,14,15);
}

// ---------------------------------------------------------------------------
// fp32 -> bf16 conversion (n is a multiple of 2048)
// ---------------------------------------------------------------------------
__global__ __launch_bounds__(256)
void convert_f32_bf16_kernel(const float* __restrict__ in,
                             bf16* __restrict__ out)
{
  const size_t i = ((size_t)blockIdx.x * blockDim.x + threadIdx.x) * 8;
  v4f a = *(const v4f*)(in + i);
  v4f b = *(const v4f*)(in + i + 4);
  v8bf o;
  #pragma unroll
  for (int e = 0; e < 4; ++e) o[e] = (bf16)a[e];
  #pragma unroll
  for (int e = 0; e < 4; ++e) o[e + 4] = (bf16)b[e];
  *(v8bf*)(out + i) = o;
}

// ---------------------------------------------------------------------------
// GEMM: Y = X @ W^T  (X: MxK bf16, W: NxK bf16 row-major), M=8192, K=N=1024.
// 128x128 tile / 256-thread WG; 8 waves in 4(row) x 2(col); each wave owns a
// 32x64 = 2x4 grid of WMMA accumulators. K-tile = 64, double-buffered async.
// ---------------------------------------------------------------------------
template<bool SPLIT_HEADS, bool ADD_BIAS, bool OUT_BF16>
__global__ __launch_bounds__(256)
void gemm_xwt_kernel(const bf16* __restrict__ A,
                     const bf16* __restrict__ W,
                     const float* __restrict__ bias,
                     void* __restrict__ Yv)
{
  constexpr int K = D_MODEL;
  constexpr int N = D_MODEL;
  constexpr int KT = 64;

  __shared__ alignas(128) bf16 As[2][128][KT];   // X tile  (m, k)
  __shared__ alignas(128) bf16 Bs[2][128][KT];   // W tile  (n, k)

  const int t    = threadIdx.x;
  const int lane = t & 31;
  const int wave = t >> 5;
  const int wm   = wave & 3;          // wave row block (x32)
  const int wn   = wave >> 2;         // wave col block (x64)
  const int lr   = lane & 15;
  const int lh   = lane >> 4;

  const int row0 = blockIdx.y * 128;
  const int col0 = blockIdx.x * 128;

  v8f acc[2][4] = {};

  // Tile copy: 128x64 bf16 = 16KB = 1024 x 16B chunks; 4 per thread per matrix.
  auto load_tile = [&](int buf, int k0) {
    #pragma unroll
    for (int p = 0; p < 4; ++p) {
      const int c  = t + 256 * p;
      const int r  = c >> 3;
      const int cc = (c & 7) * 8;
      lds_copy16(A + (size_t)(row0 + r) * K + k0 + cc, &As[buf][r][cc]);
      lds_copy16(W + (size_t)(col0 + r) * K + k0 + cc, &Bs[buf][r][cc]);
    }
  };

  load_tile(0, 0);

  for (int it = 0; it < K / KT; ++it) {
    const int cur = it & 1;
    const bool has_next = (it + 1) < (K / KT);
    if (has_next) {
      load_tile(cur ^ 1, (it + 1) * KT);
      wait_async_lds<8>();            // this tile's 8 async chunks landed
    } else {
      wait_async_lds<0>();
    }
    __syncthreads();

    #pragma unroll
    for (int kk = 0; kk < KT; kk += 32) {
      // A (16x32): lane m = l&15; K elems = lh*16 + e (contiguous 16)
      v16bf afrag[2];
      #pragma unroll
      for (int i = 0; i < 2; ++i) {
        const int m = wm * 32 + i * 16 + lr;
        afrag[i] = *(const v16bf*)&As[cur][m][kk + lh * 16];
      }
      // B (32x16): lane n = l&15; K elems = lh*16 + e (contiguous 16)
      v16bf bfrag[4];
      #pragma unroll
      for (int j = 0; j < 4; ++j) {
        const int n = wn * 64 + j * 16 + lr;
        bfrag[j] = *(const v16bf*)&Bs[cur][n][kk + lh * 16];
      }
      #pragma unroll
      for (int i = 0; i < 2; ++i)
        #pragma unroll
        for (int j = 0; j < 4; ++j)
          acc[i][j] = wmma_bf16(afrag[i], bfrag[j], acc[i][j]);
    }
    __syncthreads();
  }

  // --- epilogue: C/D layout row m = r + lh*8, col n = lr ------------------
  #pragma unroll
  for (int i = 0; i < 2; ++i) {
    #pragma unroll
    for (int j = 0; j < 4; ++j) {
      const int col = col0 + wn * 64 + j * 16 + lr;
      float bv = 0.0f;
      if constexpr (ADD_BIAS) bv = bias[col];
      #pragma unroll
      for (int r = 0; r < 8; ++r) {
        const int row = row0 + wm * 32 + i * 16 + lh * 8 + r;
        float v = acc[i][j][r];
        if constexpr (ADD_BIAS) v += bv;
        size_t off;
        if constexpr (SPLIT_HEADS) {
          // (b*S+s, h*64+d) -> (B,H,S,Dh)
          const int b = row >> 11, s = row & (SEQ - 1);
          const int h = col >> 6,  d = col & (D_HEAD - 1);
          off = ((((size_t)b * N_HEADS + h) * SEQ + s) << 6) + d;
        } else {
          off = (size_t)row * N + col;
        }
        if constexpr (OUT_BF16) ((bf16*)Yv)[off] = (bf16)v;
        else                    ((float*)Yv)[off] = v;
      }
    }
  }
}

// ---------------------------------------------------------------------------
// Causal flash attention. One WG = (b,h) x 128 q-rows; 4 waves x 32 q-rows.
// K tiles of 64 keys streamed via async-to-LDS; V transposed into LDS via
// regular loads (overlaps with async); online softmax; P spilled to LDS in
// bf16 and re-read as WMMA A-fragments (per-wave region, no barrier needed).
// ---------------------------------------------------------------------------
__global__ __launch_bounds__(128)
void flash_attn_kernel(const bf16* __restrict__ Q,
                       const bf16* __restrict__ Kh_,
                       const bf16* __restrict__ Vh_,
                       bf16* __restrict__ Ohead)   // (B, S, H*Dh)
{
  const int bh   = blockIdx.y;                // 0..B*H-1
  const int q0   = blockIdx.x * 128;
  const int t    = threadIdx.x;
  const int lane = t & 31;
  const int wave = t >> 5;
  const int lr   = lane & 15;
  const int lh   = lane >> 4;

  const bf16* Qh = Q   + (size_t)bh * SEQ * D_HEAD;
  const bf16* Kh = Kh_ + (size_t)bh * SEQ * D_HEAD;
  const bf16* Vh = Vh_ + (size_t)bh * SEQ * D_HEAD;

  __shared__ alignas(128) bf16 Ks[64][64];    // (key, d)
  __shared__ alignas(128) bf16 Vs[64][64];    // transposed: (d, key)
  __shared__ alignas(128) bf16 Ps[128][64];   // probs (q, key)

  // Q fragments held in registers for the whole kernel: [row frag][k-step]
  v16bf qf[2][2];
  #pragma unroll
  for (int i = 0; i < 2; ++i) {
    const int qrow = q0 + wave * 32 + i * 16 + lr;
    const bf16* qp = Qh + (size_t)qrow * D_HEAD;
    #pragma unroll
    for (int ks = 0; ks < 2; ++ks) {
      qf[i][ks] = *(const v16bf*)(qp + ks * 32 + lh * 16);
    }
  }

  v8f o[2][4] = {};
  float rmax[2][8], rsum[2][8];
  #pragma unroll
  for (int i = 0; i < 2; ++i)
    #pragma unroll
    for (int r = 0; r < 8; ++r) { rmax[i][r] = -3.0e38f; rsum[i][r] = 0.0f; }

  const float scale = 0.125f;                 // 1/sqrt(Dh)
  const int kend = q0 + 128;                  // causal upper bound (exclusive)

  for (int j = 0; j < kend; j += 64) {
    // --- K tile via async-to-LDS: 64x64 bf16 = 512 x 16B; 4 chunks/thread --
    #pragma unroll
    for (int p = 0; p < 4; ++p) {
      const int c  = t + 128 * p;
      const int r  = c >> 3;
      const int cc = (c & 7) * 8;
      lds_copy16(Kh + (size_t)(j + r) * D_HEAD + cc, &Ks[r][cc]);
    }
    // --- V tile transposed via regular loads (overlaps the async K copy) ---
    {
      const int r  = t >> 1;                  // key row 0..63
      const int c0 = (t & 1) * 32;            // d half
      const bf16* vp = Vh + (size_t)(j + r) * D_HEAD + c0;
      #pragma unroll
      for (int p = 0; p < 4; ++p) {
        v8bf vv = *(const v8bf*)(vp + p * 8);
        #pragma unroll
        for (int e = 0; e < 8; ++e) Vs[c0 + p * 8 + e][r] = vv[e];
      }
    }
    wait_async_lds<0>();
    __syncthreads();

    // --- S = Q K^T --------------------------------------------------------
    v8f sc[2][4] = {};
    #pragma unroll
    for (int ks = 0; ks < 2; ++ks) {
      v16bf bf[4];
      #pragma unroll
      for (int n = 0; n < 4; ++n)
        bf[n] = *(const v16bf*)&Ks[n * 16 + lr][ks * 32 + lh * 16];
      #pragma unroll
      for (int i = 0; i < 2; ++i)
        #pragma unroll
        for (int n = 0; n < 4; ++n)
          sc[i][n] = wmma_bf16(qf[i][ks], bf[n], sc[i][n]);
    }

    // --- scale + causal mask + online softmax -----------------------------
    #pragma unroll
    for (int i = 0; i < 2; ++i) {
      #pragma unroll
      for (int n = 0; n < 4; ++n) {
        const int kidx = j + n * 16 + lr;
        #pragma unroll
        for (int r = 0; r < 8; ++r) {
          const int qidx = q0 + wave * 32 + i * 16 + lh * 8 + r;
          const float v = sc[i][n][r] * scale;
          sc[i][n][r] = (kidx <= qidx) ? v : -1.0e10f;
        }
      }
      #pragma unroll
      for (int r = 0; r < 8; ++r) {
        float m = sc[i][0][r];
        #pragma unroll
        for (int n = 1; n < 4; ++n) m = fmaxf(m, sc[i][n][r]);
        #pragma unroll
        for (int off = 1; off < 16; off <<= 1)
          m = fmaxf(m, __shfl_xor(m, off, 32));
        const float mprev = rmax[i][r];
        const float mn    = fmaxf(mprev, m);
        rmax[i][r] = mn;
        const float corr = __expf(mprev - mn);
        rsum[i][r] *= corr;
        #pragma unroll
        for (int d = 0; d < 4; ++d) o[i][d][r] *= corr;
        float psum = 0.0f;
        #pragma unroll
        for (int n = 0; n < 4; ++n) {
          const float p = __expf(sc[i][n][r] - mn);
          sc[i][n][r] = p;
          psum += p;
        }
        #pragma unroll
        for (int off = 1; off < 16; off <<= 1)
          psum += __shfl_xor(psum, off, 32);
        rsum[i][r] += psum;
      }
      // spill P (bf16) into this wave's private LDS region
      #pragma unroll
      for (int n = 0; n < 4; ++n)
        #pragma unroll
        for (int r = 0; r < 8; ++r)
          Ps[wave * 32 + i * 16 + lh * 8 + r][n * 16 + lr] = (bf16)sc[i][n][r];
    }

    // --- O += P V ---------------------------------------------------------
    #pragma unroll
    for (int ks = 0; ks < 2; ++ks) {
      v16bf pf[2];
      #pragma unroll
      for (int i = 0; i < 2; ++i) {
        const int m = wave * 32 + i * 16 + lr;
        pf[i] = *(const v16bf*)&Ps[m][ks * 32 + lh * 16];
      }
      v16bf vf[4];
      #pragma unroll
      for (int d = 0; d < 4; ++d)
        vf[d] = *(const v16bf*)&Vs[d * 16 + lr][ks * 32 + lh * 16];
      #pragma unroll
      for (int i = 0; i < 2; ++i)
        #pragma unroll
        for (int d = 0; d < 4; ++d)
          o[i][d] = wmma_bf16(pf[i], vf[d], o[i][d]);
    }
    __syncthreads();
  }

  // --- normalize + write heads_out as (B, S, H*Dh) bf16 -------------------
  const int b = bh >> 4, h = bh & (N_HEADS - 1);
  #pragma unroll
  for (int i = 0; i < 2; ++i) {
    #pragma unroll
    for (int d4 = 0; d4 < 4; ++d4) {
      const int d = d4 * 16 + lr;
      #pragma unroll
      for (int r = 0; r < 8; ++r) {
        const int q = q0 + wave * 32 + i * 16 + lh * 8 + r;
        const float val = o[i][d4][r] / rsum[i][r];
        Ohead[((size_t)(b * SEQ + q) * D_MODEL) + h * D_HEAD + d] = (bf16)val;
      }
    }
  }
}

// ---------------------------------------------------------------------------
// Launch: convert -> QKV projections -> flash attention -> output projection.
// Workspace (bf16): Xq,Xk,Xv (3x16MB) | Wq,Wk,Wv,Wp (4x2MB) | Q,K,V (3x16MB)
//                   | heads (16MB)  => 120MB total.
// ---------------------------------------------------------------------------
extern "C" void kernel_launch(void* const* d_in, const int* in_sizes, int n_in,
                              void* d_out, int out_size, void* d_ws, size_t ws_size,
                              hipStream_t stream) {
  const float* queries = (const float*)d_in[0];
  const float* keys    = (const float*)d_in[1];
  const float* values  = (const float*)d_in[2];
  // d_in[3] is the boolean mask; causality is applied analytically.
  const float* W_q     = (const float*)d_in[4];
  const float* W_k     = (const float*)d_in[5];
  const float* W_v     = (const float*)d_in[6];
  const float* W_proj  = (const float*)d_in[7];
  const float* b_proj  = (const float*)d_in[8];

  const size_t act = (size_t)MROWS * D_MODEL;     // 8M elements
  const size_t wgt = (size_t)D_MODEL * D_MODEL;   // 1M elements

  bf16* Xq = (bf16*)d_ws;
  bf16* Xk = Xq + act;
  bf16* Xv = Xk + act;
  bf16* Wq = Xv + act;
  bf16* Wk = Wq + wgt;
  bf16* Wv = Wk + wgt;
  bf16* Wp = Wv + wgt;
  bf16* Qb = Wp + wgt;
  bf16* Kb = Qb + act;
  bf16* Vb = Kb + act;
  bf16* Hb = Vb + act;

  // --- pass 0: fp32 -> bf16 ------------------------------------------------
  const int cb = 256;
  convert_f32_bf16_kernel<<<dim3(act / (8 * cb)), cb, 0, stream>>>(queries, Xq);
  convert_f32_bf16_kernel<<<dim3(act / (8 * cb)), cb, 0, stream>>>(keys,    Xk);
  convert_f32_bf16_kernel<<<dim3(act / (8 * cb)), cb, 0, stream>>>(values,  Xv);
  convert_f32_bf16_kernel<<<dim3(wgt / (8 * cb)), cb, 0, stream>>>(W_q,    Wq);
  convert_f32_bf16_kernel<<<dim3(wgt / (8 * cb)), cb, 0, stream>>>(W_k,    Wk);
  convert_f32_bf16_kernel<<<dim3(wgt / (8 * cb)), cb, 0, stream>>>(W_v,    Wv);
  convert_f32_bf16_kernel<<<dim3(wgt / (8 * cb)), cb, 0, stream>>>(W_proj, Wp);

  dim3 ggrid(D_MODEL / 128, MROWS / 128);         // 8 x 64
  dim3 gblock(256);

  // --- pass 1: QKV projections (split-heads, bf16 out) ---------------------
  gemm_xwt_kernel<true,  false, true ><<<ggrid, gblock, 0, stream>>>(Xq, Wq, nullptr, Qb);
  gemm_xwt_kernel<true,  false, true ><<<ggrid, gblock, 0, stream>>>(Xk, Wk, nullptr, Kb);
  gemm_xwt_kernel<true,  false, true ><<<ggrid, gblock, 0, stream>>>(Xv, Wv, nullptr, Vb);

  // --- pass 2: causal flash attention --------------------------------------
  dim3 fgrid(SEQ / 128, BATCH * N_HEADS);         // 16 x 64
  flash_attn_kernel<<<fgrid, dim3(128), 0, stream>>>(Qb, Kb, Vb, Hb);

  // --- pass 3: output projection + bias, fp32 out --------------------------
  gemm_xwt_kernel<false, true,  false><<<ggrid, gblock, 0, stream>>>(Hb, Wp, b_proj, (float*)d_out);
}
